// AttentionBlock_19052474925666
// MI455X (gfx1250) — matrix operand order repaired and checked by reading, not compiled, not run on hardware
//
#include <hip/hip_runtime.h>

// ---------------------------------------------------------------------------
// AttentionBlock for MI455X (gfx1250): GroupNorm -> QKV GEMM -> flash
// attention -> proj GEMM + residual.  Matmuls use v_wmma_f32_16x16x32_bf16
// (fp32 accum); tiles staged global->LDS by the Tensor Data Mover with D#
// padding for 16B-aligned pitches; all WMMA fragments load as ds_load_b128.
//
// Layouts:
//   A operands: row-major [row][K] bf16  -> A frag = 2x b128 per lane
//   B operands: "BPK" [col][K/2] dwords (dword = bf16 pair along K)
//               -> B frag = two 4-dword runs 8 dwords apart per lane = 2x b128
// ---------------------------------------------------------------------------

#define Bn   4
#define Cc   256
#define Nn   4096
#define NHh  4
#define HDd  64
#define CPGg 32      // channels per group (256/8)
#define EPSf 1e-5f

typedef __bf16        v16bf __attribute__((ext_vector_type(16)));
typedef float         v8f   __attribute__((ext_vector_type(8)));
typedef unsigned int  u32x4 __attribute__((ext_vector_type(4)));
typedef int           i32x8 __attribute__((ext_vector_type(8)));
typedef int           i32x4 __attribute__((ext_vector_type(4)));

union ABfrag { v16bf v; unsigned u[8]; };

__device__ __forceinline__ unsigned short f2bf(float x) {
  unsigned u = __float_as_uint(x);
  u += 0x7FFFu + ((u >> 16) & 1u);          // round-to-nearest-even
  return (unsigned short)(u >> 16);
}

__device__ __forceinline__ v8f zero8() {
  v8f z;
#pragma unroll
  for (int r = 0; r < 8; ++r) z[r] = 0.0f;
  return z;
}

__device__ __forceinline__ v8f wmma_bf16(v16bf a, v16bf b, v8f c) {
  return __builtin_amdgcn_wmma_f32_16x16x32_bf16(false, a, false, b, (short)0, c,
                                                 false, false);
}

// A fragment: 16x32 bf16 from row-major LDS (pitch ld u16, 16B-aligned rows).
// Lane (m,kh): VGPR0-3 = K 8kh..8kh+7, VGPR4-7 = K 16+8kh..16+8kh+7 -> 2x b128.
__device__ __forceinline__ v16bf load_a_frag(const unsigned short* base, int ld,
                                             int row0, int k0) {
  const int lane = threadIdx.x & 31;
  const int m = lane & 15, kh = lane >> 4;
  const unsigned* p = (const unsigned*)(base + (row0 + m) * ld + k0 + 8 * kh);
  const uint4 lo = *(const uint4*)p;
  const uint4 hi = *(const uint4*)(p + 8);
  ABfrag f;
  f.u[0] = lo.x; f.u[1] = lo.y; f.u[2] = lo.z; f.u[3] = lo.w;
  f.u[4] = hi.x; f.u[5] = hi.y; f.u[6] = hi.z; f.u[7] = hi.w;
  return f.v;
}

// B fragment from BPK LDS (dword array, pitch ldd dwords, 16B-aligned rows).
// Lane (n,kh): dwords kp0+4kh..+3 and kp0+8+4kh..+3 (K 8kh.., 16+8kh..) -> 2x b128.
__device__ __forceinline__ v16bf load_b_pk(const unsigned* based, int ldd,
                                           int kp0, int n0) {
  const int lane = threadIdx.x & 31;
  const unsigned* p = based + (n0 + (lane & 15)) * ldd + kp0 + 4 * (lane >> 4);
  const uint4 lo = *(const uint4*)p;
  const uint4 hi = *(const uint4*)(p + 8);
  ABfrag f;
  f.u[0] = lo.x; f.u[1] = lo.y; f.u[2] = lo.z; f.u[3] = lo.w;
  f.u[4] = hi.x; f.u[5] = hi.y; f.u[6] = hi.z; f.u[7] = hi.w;
  return f.v;
}

// TDM 2D tile load.  tile_d0/d1 and stride0 in elements of size 2^ds_code
// bytes (1 = 2B bf16, 2 = 4B dword).  LDS rows padded by (pad_amt+1) dwords
// every 2^(pad_int+1) dwords.
__device__ __forceinline__ void tdm_load_2d(void* ldsptr, const void* gsrc,
                                            unsigned tile_d0, unsigned tile_d1,
                                            unsigned long long stride0,
                                            unsigned pad_int, unsigned pad_amt,
                                            unsigned ds_code) {
#if __has_builtin(__builtin_amdgcn_tensor_load_to_lds)
  const unsigned long long ga = (unsigned long long)gsrc;
  const unsigned ldsoff = (unsigned)(unsigned long long)ldsptr;
  u32x4 g0; i32x8 g1; i32x4 g2, g3; i32x8 g4;
  g0[0] = 1u;                                          // count=1
  g0[1] = ldsoff;                                      // lds_addr
  g0[2] = (unsigned)ga;                                // global_addr lo
  g0[3] = (unsigned)((ga >> 32) & 0x1FFFFFFull) | (2u << 30);  // addr hi | type=2
  g1[0] = (int)((ds_code << 16) | (1u << 20) | (pad_int << 22) | (pad_amt << 25));
  g1[1] = (int)((tile_d0 & 0xFFFFu) << 16);            // tensor_dim0 = tile_d0
  g1[2] = (int)((tile_d1 & 0xFFFFu) << 16);            // tensor_dim1 = tile_d1
  g1[3] = (int)((tile_d0 & 0xFFFFu) << 16);            // tile_dim0
  g1[4] = (int)(tile_d1 & 0xFFFFu);                    // tile_dim1 (tile_dim2=0)
  g1[5] = (int)(stride0 & 0xFFFFFFFFull);              // dim0 stride lo
  g1[6] = (int)((stride0 >> 32) & 0xFFFFull);          // dim0 stride hi
  g1[7] = 0;
  g2[0] = g2[1] = g2[2] = g2[3] = 0;
  g3[0] = g3[1] = g3[2] = g3[3] = 0;
#pragma unroll
  for (int i = 0; i < 8; ++i) g4[i] = 0;
  __builtin_amdgcn_tensor_load_to_lds(g0, g1, g2, g3, g4, 0);
#else
  const int lane = threadIdx.x & 31;
  const unsigned padfac = (ds_code == 2) ? 1u : 2u;     // pad dwords -> elements
  const unsigned pitch = tile_d0 + (pad_amt + 1u) * padfac;
  if (ds_code == 2) {
    unsigned* d = (unsigned*)ldsptr;
    const unsigned* s = (const unsigned*)gsrc;
    for (unsigned row = 0; row < tile_d1; ++row)
      for (unsigned i = lane; i < tile_d0; i += 32)
        d[row * pitch + i] = s[row * stride0 + i];
  } else {
    unsigned short* d = (unsigned short*)ldsptr;
    const unsigned short* s = (const unsigned short*)gsrc;
    for (unsigned row = 0; row < tile_d1; ++row)
      for (unsigned i = lane; i < tile_d0; i += 32)
        d[row * pitch + i] = s[row * stride0 + i];
  }
#endif
}

__device__ __forceinline__ void tdm_wait0() {
#if __has_builtin(__builtin_amdgcn_s_wait_tensorcnt)
  __builtin_amdgcn_s_wait_tensorcnt(0);
#endif
}

// ---------------------------------------------------------------------------
// prep kernels
// ---------------------------------------------------------------------------
__global__ __launch_bounds__(256) void cvt_f32_bf16(const float* __restrict__ src,
                                                    unsigned short* __restrict__ dst,
                                                    int n) {
  const int i = blockIdx.x * 256 + threadIdx.x;
  if (i < n) dst[i] = f2bf(src[i]);
}

__global__ __launch_bounds__(256) void gn_stats(const float* __restrict__ x,
                                                float* __restrict__ stats) {
  __shared__ float rs[256], rq[256];
  const int tid = threadIdx.x;
  const int bg = blockIdx.x;                 // b*8 + g
  const float* base = x + (size_t)bg * CPGg * Nn;
  float s = 0.f, q = 0.f;
  for (int i = tid; i < CPGg * Nn; i += 256) {
    const float v = base[i];
    s += v; q += v * v;
  }
  rs[tid] = s; rq[tid] = q;
  __syncthreads();
  for (int st = 128; st > 0; st >>= 1) {
    if (tid < st) { rs[tid] += rs[tid + st]; rq[tid] += rq[tid + st]; }
    __syncthreads();
  }
  if (tid == 0) {
    const float inv = 1.0f / (float)(CPGg * Nn);
    const float mean = rs[0] * inv;
    const float var  = rq[0] * inv - mean * mean;
    stats[bg * 2]     = mean;
    stats[bg * 2 + 1] = rsqrtf(var + EPSf);
  }
}

// GroupNorm apply + transpose into hpk[b][n][c/2] (BPK for the QKV GEMM).
// 64x64 (c,n) tile through LDS: both global sides coalesced.
__global__ __launch_bounds__(256) void gn_apply_t(const float* __restrict__ x,
                                                  const float* __restrict__ stats,
                                                  const float* __restrict__ gw,
                                                  const float* __restrict__ gb,
                                                  unsigned* __restrict__ hpk) {
  __shared__ float xs[64][65];
  const int tid = threadIdx.x;
  const int n0 = blockIdx.x * 64, c0 = blockIdx.y * 64, b = blockIdx.z;
  for (int idx = tid; idx < 64 * 64; idx += 256) {
    const int cl = idx >> 6, nl = idx & 63;
    const int c = c0 + cl;
    const int bg = b * 8 + (c >> 5);
    const float mean = stats[bg * 2], rstd = stats[bg * 2 + 1];
    const float v = x[((size_t)b * Cc + c) * Nn + n0 + nl];
    xs[cl][nl] = (v - mean) * rstd * gw[c] + gb[c];
  }
  __syncthreads();
  for (int idx = tid; idx < 64 * 32; idx += 256) {
    const int pcl = idx & 31, nl = idx >> 5;
    const unsigned lo = f2bf(xs[2 * pcl][nl]);
    const unsigned hi = f2bf(xs[2 * pcl + 1][nl]);
    hpk[((size_t)b * Nn + n0 + nl) * (Cc / 2) + (c0 >> 1) + pcl] = lo | (hi << 16);
  }
}

// ---------------------------------------------------------------------------
// WMMA GEMM:  OUT[o][n] = sum_c A[o][c] * B[c][n]  (per batch)
// A: bf16 row-major [Mtot][256]; B: BPK dwords [n][128].
// Block tile 64x64, K-step 32, 8 waves (4 row blocks x 2 col halves).
// MODE 0 = QKV epilogue (bias; q->qT[n][d], k->kpk[m][d/2], v->v[d][m])
// MODE 1 = proj epilogue (bias + residual, fp32 out)
// ---------------------------------------------------------------------------
template <int MODE>
__global__ __launch_bounds__(256) void gemm_nn(const unsigned short* __restrict__ A,
                                               const unsigned* __restrict__ Bpk,
                                               const float* __restrict__ bias,
                                               const float* __restrict__ xres,
                                               float* __restrict__ outf,
                                               unsigned short* __restrict__ qT,
                                               unsigned* __restrict__ kpk,
                                               unsigned short* __restrict__ vB) {
  __shared__ unsigned short As[64 * 40];   // 64 rows x 32 K u16, pitch 40
  __shared__ unsigned Bsd[64 * 20];        // 64 n-rows x 16 K-pair dwords, pitch 20
  __shared__ unsigned short Cs[64 * 72];

  const int tid = threadIdx.x;
  const int w = tid >> 5, lane = tid & 31;
  const int nt = blockIdx.x, mt = blockIdx.y, b = blockIdx.z;
  const int n0 = nt * 64, m0 = mt * 64;
  const unsigned* Bm = Bpk + (size_t)b * Nn * (Cc / 2);
  const int wr = (w >> 1) * 16;   // wave row block
  const int wc = (w & 1) * 32;    // wave col half

  v8f acc[2];
  acc[0] = zero8(); acc[1] = zero8();

  for (int ks = 0; ks < Cc / 32; ++ks) {
    if (tid < 32) {
      // A tile: u16, rows 32 elems (16 dw) -> pad codes (3,3) -> pitch 40 u16
      tdm_load_2d(As, A + (size_t)m0 * Cc + ks * 32, 32, 64,
                  (unsigned long long)Cc, 3u, 3u, 1u);
      // B tile: dwords, rows 16 dw -> pad codes (3,3) -> pitch 20 dw
      tdm_load_2d(Bsd, Bm + (size_t)n0 * (Cc / 2) + ks * 16, 16, 64,
                  (unsigned long long)(Cc / 2), 3u, 3u, 2u);
      tdm_wait0();
    }
    __syncthreads();
    const v16bf a = load_a_frag(As, 40, wr, 0);
#pragma unroll
    for (int t = 0; t < 2; ++t)
      acc[t] = wmma_bf16(a, load_b_pk(Bsd, 20, 0, wc + t * 16), acc[t]);
    __syncthreads();
  }

  const int colL = lane & 15;
  const int rad = (lane >> 4) * 8;        // D layout: lanes 16-31 hold rows M+8

  if (MODE == 1) {
#pragma unroll
    for (int t = 0; t < 2; ++t)
#pragma unroll
      for (int r = 0; r < 8; ++r) {
        const int row = wr + r + rad;
        const int colg = wc + t * 16 + colL;
        const int o = m0 + row;
        const size_t gi = ((size_t)b * Cc + o) * Nn + n0 + colg;
        outf[gi] = acc[t][r] + bias[o] + xres[gi];
      }
  } else {
#pragma unroll
    for (int t = 0; t < 2; ++t)
#pragma unroll
      for (int r = 0; r < 8; ++r) {
        const int row = wr + r + rad;
        const int colg = wc + t * 16 + colL;
        Cs[row * 72 + colg] = f2bf(acc[t][r] + bias[m0 + row]);
      }
    __syncthreads();
    const int sec = mt >> 2;              // 0=q 1=k 2=v  (64 | head dim)
    const int head = mt & 3;
    const int bh = b * NHh + head;
    if (sec == 0) {                       // q -> qT[n][d]
      for (int idx = tid; idx < 64 * 64; idx += 256) {
        const int nl = idx >> 6, d = idx & 63;
        qT[((size_t)bh * Nn + n0 + nl) * HDd + d] = Cs[d * 72 + nl];
      }
    } else if (sec == 1) {                // k -> kpk[m][d/2] (BPK for S GEMM)
      for (int idx = tid; idx < 64 * 32; idx += 256) {
        const int pd = idx & 31, ml = idx >> 5;
        const unsigned lo = Cs[(2 * pd) * 72 + ml];
        const unsigned hi = Cs[(2 * pd + 1) * 72 + ml];
        kpk[((size_t)bh * Nn + n0 + ml) * (HDd / 2) + pd] = lo | (hi << 16);
      }
    } else {                              // v stays [d][m] (dword pairs = BPK)
      for (int idx = tid; idx < 64 * 64; idx += 256) {
        const int row = idx >> 6, cl = idx & 63;
        vB[((size_t)bh * HDd + row) * Nn + n0 + cl] = Cs[row * 72 + cl];
      }
    }
  }
}

// ---------------------------------------------------------------------------
// Flash attention: block = (128 q rows) x (b,head); 8 waves each own 16 rows.
// S = qT x kpk (WMMA), online softmax fp32 (wave16 shuffles), O += P x v.
// K/V 64-col tiles streamed via TDM.
// ---------------------------------------------------------------------------
__global__ __launch_bounds__(256) void flash_attn(const unsigned short* __restrict__ qT,
                                                  const unsigned* __restrict__ kpk,
                                                  const unsigned short* __restrict__ vB,
                                                  unsigned* __restrict__ opk) {
  __shared__ unsigned short Qs[128 * 72];  // q rows [n][d], pitch 72 u16
  __shared__ unsigned Ksd[64 * 36];        // BPK [m][d/2], pitch 36 dw
  __shared__ unsigned Vsd[64 * 36];        // v [d][m] u16 = BPK dwords, pitch 36 dw
  __shared__ unsigned short Ps[128 * 72];  // P rows [n][m], pitch 72 u16

  const int tid = threadIdx.x;
  const int w = tid >> 5, lane = tid & 31;
  const int bh = blockIdx.y;              // b*4 + h
  const int nt = blockIdx.x;              // 128-row q tile
  const float scale = 0.125f;             // 64^-0.5

  if (tid < 32)
    tdm_load_2d(Qs, qT + ((size_t)bh * Nn + nt * 128) * HDd, 64, 128, 64ull,
                4u, 3u, 1u);

  v8f o[4];
  float m_i[8], l_i[8];
#pragma unroll
  for (int t = 0; t < 4; ++t) o[t] = zero8();
#pragma unroll
  for (int r = 0; r < 8; ++r) { m_i[r] = -3.0e38f; l_i[r] = 0.0f; }

  const int colL = lane & 15;
  const int rad = (lane >> 4) * 8;

  for (int j = 0; j < Nn / 64; ++j) {
    if (tid < 32) {
      // K tile: BPK dwords, rows m=64 x 32 dw, dense -> pad (4,3) -> pitch 36
      tdm_load_2d(Ksd, kpk + ((size_t)bh * Nn + j * 64) * (HDd / 2), 32, 64,
                  (unsigned long long)(HDd / 2), 4u, 3u, 2u);
      // V tile: u16 rows d=64 x 64 elems, stride N -> pad (4,3) -> pitch 72 u16
      tdm_load_2d(Vsd, vB + (size_t)bh * HDd * Nn + j * 64, 64, 64,
                  (unsigned long long)Nn, 4u, 3u, 1u);
      tdm_wait0();
    }
    __syncthreads();

    // S tile: my 16 rows x 64 cols, K-dim = 64
    const v16bf qa0 = load_a_frag(Qs, 72, w * 16, 0);
    const v16bf qa1 = load_a_frag(Qs, 72, w * 16, 32);
    v8f s[4];
#pragma unroll
    for (int t = 0; t < 4; ++t) {
      v8f a = zero8();
      a = wmma_bf16(qa0, load_b_pk(Ksd, 36, 0,  t * 16), a);
      a = wmma_bf16(qa1, load_b_pk(Ksd, 36, 16, t * 16), a);
      s[t] = a;
    }

    // online softmax per row (row r lives in 16 lanes of one half-wave)
#pragma unroll
    for (int r = 0; r < 8; ++r) {
      float a0 = s[0][r] * scale, a1 = s[1][r] * scale;
      float a2 = s[2][r] * scale, a3 = s[3][r] * scale;
      float mx = fmaxf(fmaxf(a0, a1), fmaxf(a2, a3));
      mx = fmaxf(mx, __shfl_xor(mx, 1, 16));
      mx = fmaxf(mx, __shfl_xor(mx, 2, 16));
      mx = fmaxf(mx, __shfl_xor(mx, 4, 16));
      mx = fmaxf(mx, __shfl_xor(mx, 8, 16));
      const float mnew = fmaxf(m_i[r], mx);
      const float al = __expf(m_i[r] - mnew);
      a0 = __expf(a0 - mnew); a1 = __expf(a1 - mnew);
      a2 = __expf(a2 - mnew); a3 = __expf(a3 - mnew);
      float rs = a0 + a1 + a2 + a3;
      rs += __shfl_xor(rs, 1, 16);
      rs += __shfl_xor(rs, 2, 16);
      rs += __shfl_xor(rs, 4, 16);
      rs += __shfl_xor(rs, 8, 16);
      l_i[r] = l_i[r] * al + rs;
      m_i[r] = mnew;
      o[0][r] *= al; o[1][r] *= al; o[2][r] *= al; o[3][r] *= al;
      const int prow = (w * 16 + r + rad) * 72;
      Ps[prow + 0 * 16 + colL] = f2bf(a0);
      Ps[prow + 1 * 16 + colL] = f2bf(a1);
      Ps[prow + 2 * 16 + colL] = f2bf(a2);
      Ps[prow + 3 * 16 + colL] = f2bf(a3);
    }

    // O += P x V (wave reads only its own P rows; same-wave LDS is ordered)
    const v16bf pa0 = load_a_frag(Ps, 72, w * 16, 0);
    const v16bf pa1 = load_a_frag(Ps, 72, w * 16, 32);
#pragma unroll
    for (int t = 0; t < 4; ++t) {
      v8f a = o[t];
      a = wmma_bf16(pa0, load_b_pk(Vsd, 36, 0,  t * 16), a);
      a = wmma_bf16(pa1, load_b_pk(Vsd, 36, 16, t * 16), a);
      o[t] = a;
    }
    __syncthreads();   // protect Ks/Vs before next TDM
  }

  // finalize: O /= l, bounce through LDS (reuse Qs as [n][d]), emit opk dwords
#pragma unroll
  for (int r = 0; r < 8; ++r) {
    const float inv = 1.0f / l_i[r];
    const int qrow = (w * 16 + r + rad) * 72;
#pragma unroll
    for (int t = 0; t < 4; ++t)
      Qs[qrow + t * 16 + colL] = f2bf(o[t][r] * inv);
  }
  __syncthreads();
  const int b = bh >> 2, h = bh & 3;
  const unsigned* Qsd = (const unsigned*)Qs;           // [n][d/2], pitch 36 dw
  for (int idx = tid; idx < 128 * 32; idx += 256) {
    const int pd = idx & 31, nl = idx >> 5;
    opk[((size_t)b * Nn + nt * 128 + nl) * (Cc / 2) + h * 32 + pd] =
        Qsd[nl * 36 + pd];
  }
}

// ---------------------------------------------------------------------------
extern "C" void kernel_launch(void* const* d_in, const int* in_sizes, int n_in,
                              void* d_out, int out_size, void* d_ws, size_t ws_size,
                              hipStream_t stream) {
  (void)in_sizes; (void)n_in; (void)out_size; (void)ws_size;
  const float* x      = (const float*)d_in[0];
  const float* norm_w = (const float*)d_in[1];
  const float* norm_b = (const float*)d_in[2];
  const float* qkv_w  = (const float*)d_in[3];
  const float* qkv_b  = (const float*)d_in[4];
  const float* proj_w = (const float*)d_in[5];
  const float* proj_b = (const float*)d_in[6];
  float* out = (float*)d_out;

  char* ws = (char*)d_ws;
  size_t off = 0;
  float* stats = (float*)(ws + off);                     off += 256;
  unsigned short* qkvw_bf = (unsigned short*)(ws + off); off += (size_t)3 * Cc * Cc * 2;
  unsigned short* projw_bf = (unsigned short*)(ws + off); off += (size_t)Cc * Cc * 2;
  unsigned* hpk = (unsigned*)(ws + off);                 off += (size_t)Bn * Nn * (Cc / 2) * 4;
  unsigned short* qT = (unsigned short*)(ws + off);      off += (size_t)Bn * Cc * Nn * 2;
  unsigned* kpk = (unsigned*)(ws + off);                 off += (size_t)Bn * NHh * Nn * (HDd / 2) * 4;
  unsigned short* vB = (unsigned short*)(ws + off);      off += (size_t)Bn * Cc * Nn * 2;
  unsigned* opk = (unsigned*)(ws + off);                 off += (size_t)Bn * Nn * (Cc / 2) * 4;

  cvt_f32_bf16<<<(3 * Cc * Cc + 255) / 256, 256, 0, stream>>>(qkv_w, qkvw_bf, 3 * Cc * Cc);
  cvt_f32_bf16<<<(Cc * Cc + 255) / 256, 256, 0, stream>>>(proj_w, projw_bf, Cc * Cc);
  gn_stats<<<Bn * 8, 256, 0, stream>>>(x, stats);
  gn_apply_t<<<dim3(Nn / 64, Cc / 64, Bn), 256, 0, stream>>>(x, stats, norm_w,
                                                             norm_b, hpk);
  gemm_nn<0><<<dim3(Nn / 64, (3 * Cc) / 64, Bn), 256, 0, stream>>>(
      qkvw_bf, hpk, qkv_b, nullptr, nullptr, qT, kpk, vB);
  flash_attn<<<dim3(Nn / 128, Bn * NHh), 256, 0, stream>>>(qT, kpk, vB, opk);
  gemm_nn<1><<<dim3(Nn / 64, Cc / 64, Bn), 256, 0, stream>>>(
      projw_bf, opk, proj_b, x, out, nullptr, nullptr, nullptr);
}
